// RWKV_83949430768051
// MI455X (gfx1250) — matrix operand order repaired and checked by reading, not compiled, not run on hardware
//
#include <hip/hip_runtime.h>
#include <hip/hip_bf16.h>

#define Bsz   4
#define Tlen  960
#define Cdim  2048
#define Hn    32
#define Nhd   64
#define Mrows (Bsz*Tlen)   // 3840
#define PKROW (Hn*6*64)    // 12288 floats per token row in packed scan layout

typedef __attribute__((ext_vector_type(8)))  _Float16 v8h;
typedef __attribute__((ext_vector_type(16))) _Float16 v16h;
typedef __attribute__((ext_vector_type(8)))  float    v8f;
typedef __attribute__((ext_vector_type(4)))  int      v4i;

__device__ __forceinline__ float sigmoidf_(float z){ return 1.0f/(1.0f+expf(-z)); }

// async 16B global->LDS copy (gfx1250, tracked by ASYNCcnt)
__device__ __forceinline__ void async_ld16(float* g, float* l){
  __builtin_amdgcn_global_load_async_to_lds_b128(
      (__attribute__((address_space(1))) v4i*)g,
      (__attribute__((address_space(3))) v4i*)l, 0, 0);
}

// ---------------- elementwise helpers ----------------
__global__ void k_convert_f16(const float* __restrict__ src, _Float16* __restrict__ dst, int n){
  int i = blockIdx.x*blockDim.x + threadIdx.x;
  if (i < n) dst[i] = (_Float16)src[i];
}

// src [R][Cc] f32  ->  dst [Cc][R] f16   (store weights as [N][K] for WMMA B)
__global__ void k_transpose_f16(const float* __restrict__ src, _Float16* __restrict__ dst, int R, int Cc){
  int i = blockIdx.x*blockDim.x + threadIdx.x;
  if (i < R*Cc){ int r = i / Cc, c = i % Cc; dst[(size_t)c*R + r] = (_Float16)src[i]; }
}

// token-shift mix: out = x + (x_prev - x)*mix   (f16 activation for WMMA A)
__global__ void k_mix(const float* __restrict__ x, const float* __restrict__ mix,
                      _Float16* __restrict__ out, int T, int C, int n){
  int i = blockIdx.x*blockDim.x + threadIdx.x;
  if (i >= n) return;
  int c = i % C; int m = i / C; int t = m % T;
  float xc = x[i];
  float xp = (t == 0) ? 0.0f : x[i - C];
  out[i] = (_Float16)(xc + (xp - xc)*mix[c]);
}

// ---------------- WMMA GEMM: out[m,n] = sum_k A[m,k] * Bw[n,k] ----------------
// A: [M][K] f16 row-major, Bw: [N][K] f16 row-major. Epilogue fused by mode.
// pslot >= 0 => write into packed scan layout pk[m][head][slot][j] (N must be Cdim).
enum { MODE_F32=0, MODE_F16=1, MODE_TANH_F16=2, MODE_SIG_F16=3, MODE_BIAS_SIG_F32=4, MODE_W_DECAY_F32=5 };

__device__ __forceinline__ v16h load_afrag(const _Float16* p){
  // A 16x32 layout: per lane 16 contiguous halves at k = lhi*16
  v8h lo = *(const v8h*)p;
  v8h hi = *(const v8h*)(p + 8);
  return __builtin_shufflevector(lo, hi, 0,1,2,3,4,5,6,7,8,9,10,11,12,13,14,15);
}

__device__ __forceinline__ size_t out_idx(int row, int col, int N, int pslot){
  if (pslot < 0) return (size_t)row*N + col;
  return (size_t)row*PKROW + (size_t)(col >> 6)*384 + (size_t)pslot*64 + (col & 63);
}

__device__ __forceinline__ void epi_store(float vv, size_t oi, int col,
                                          float* __restrict__ outF, _Float16* __restrict__ outH,
                                          const float* __restrict__ bias, int mode){
  switch (mode){
    case MODE_F32:        outF[oi] = vv; break;
    case MODE_F16:        outH[oi] = (_Float16)vv; break;
    case MODE_TANH_F16:   outH[oi] = (_Float16)tanhf(vv); break;
    case MODE_SIG_F16:    outH[oi] = (_Float16)sigmoidf_(vv); break;
    case MODE_BIAS_SIG_F32: outF[oi] = sigmoidf_(bias[col] + vv); break;
    case MODE_W_DECAY_F32: {
      float z = bias[col] + vv;                 // w0 + hw@w2
      float w = -log1pf(expf(-z)) - 0.5f;       // -softplus(-z) - 0.5
      outF[oi] = expf(-expf(w));                // decay used by the scan
    } break;
  }
}

// NT = compile-time number of 16-wide N-tiles per wave -> branch-free inner loop,
// accumulators stay pinned in VGPRs. 2 M-tiles per wave (A reuse xNT, B reuse x2).
template<int NT>
__global__ __launch_bounds__(256)
void k_gemm_wmma(const _Float16* __restrict__ A, const _Float16* __restrict__ Bw,
                 int M, int N, int K, int n_base,
                 float* __restrict__ outF, _Float16* __restrict__ outH,
                 const float* __restrict__ bias, int mode, int pslot)
{
  const int wave = threadIdx.x >> 5;      // 8 waves: 256 rows per block
  const int lane = threadIdx.x & 31;
  const int lrow = lane & 15;             // row within A-tile / col within B,C,D tiles
  const int lhi  = lane >> 4;             // half-wave select
  const int m0 = blockIdx.y*256 + wave*32;
  const int n0 = n_base + blockIdx.x*(NT*16);
  if (m0 >= M) return;

  const _Float16* arow0 = A + (size_t)(m0 + lrow)*K      + lhi*16;
  const _Float16* arow1 = A + (size_t)(m0 + 16 + lrow)*K + lhi*16;

  const v8f vzero = {0.f,0.f,0.f,0.f,0.f,0.f,0.f,0.f};
  v8f acc0[NT], acc1[NT];
#pragma unroll
  for (int t = 0; t < NT; ++t){ acc0[t] = vzero; acc1[t] = vzero; }

  for (int k0 = 0; k0 < K; k0 += 32){
    if (k0 + 128 < K){                                  // scalar-uniform guard
      __builtin_prefetch(arow0 + k0 + 128, 0, 3);       // global_prefetch_b8 (near-cache)
      __builtin_prefetch(arow1 + k0 + 128, 0, 3);
    }
    v16h af0 = load_afrag(arow0 + k0);
    v16h af1 = load_afrag(arow1 + k0);
#pragma unroll
    for (int t = 0; t < NT; ++t){
      // B 32x16 fragment: col = lrow, 16 consecutive k per lane (lanes 16-31: k+16)
      v16h bf = *(const v16h*)(Bw + (size_t)(n0 + t*16 + lrow)*K + k0 + lhi*16);
      acc0[t] = __builtin_amdgcn_wmma_f32_16x16x32_f16(false, af0, false, bf,
                                                       (short)0, acc0[t], false, false);
      acc1[t] = __builtin_amdgcn_wmma_f32_16x16x32_f16(false, af1, false, bf,
                                                       (short)0, acc1[t], false, false);
    }
  }

#pragma unroll
  for (int t = 0; t < NT; ++t){
    int col = n0 + t*16 + lrow;
#pragma unroll
    for (int r = 0; r < 8; ++r){
      int row = m0 + lhi*8 + r;           // D layout: lanes 0-15 rows 0-7, lanes 16-31 rows 8-15
      epi_store(acc0[t][r], out_idx(row,    col, N, pslot), col, outF, outH, bias, mode);
      epi_store(acc1[t][r], out_idx(row+16, col, N, pslot), col, outF, outH, bias, mode);
    }
  }
}

// ---------------- per-head prep on packed layout ----------------
// pk slots per (token,head): 0=decay 1=q(r) 2=kraw->k_final 3=a->-kk 4=vsig->kk*a 5=vraw->v_final
__global__ __launch_bounds__(256)
void k_prep(float* __restrict__ pk, const float* __restrict__ v_first,
            const float* __restrict__ k_k, const float* __restrict__ k_a, int heads)
{
  int w = blockIdx.x*8 + (threadIdx.x >> 5);
  int lane = threadIdx.x & 31;
  if (w >= heads) return;
  int m = w / Hn, h = w % Hn;
  size_t bp = (size_t)w*384;
  size_t vb = (size_t)m*Cdim + h*Nhd;
  int c0 = h*Nhd + lane, c1 = c0 + 32;

  float kr0 = pk[bp+128+lane], kr1 = pk[bp+160+lane];
  float a0  = pk[bp+192+lane], a1  = pk[bp+224+lane];
  float vs0 = pk[bp+256+lane], vs1 = pk[bp+288+lane];
  float vr0 = pk[bp+320+lane], vr1 = pk[bp+352+lane];
  float vf0 = v_first[vb+lane], vf1 = v_first[vb+32+lane];

  float kk0 = kr0*k_k[c0], kk1 = kr1*k_k[c1];
  float ss = kk0*kk0 + kk1*kk1;
#pragma unroll
  for (int mm = 16; mm >= 1; mm >>= 1) ss += __shfl_xor(ss, mm, 32);
  float inv = 1.0f / fmaxf(sqrtf(ss), 1e-12f);
  float kn0 = kk0*inv, kn1 = kk1*inv;

  pk[bp+128+lane] = kr0*(1.0f + (a0 - 1.0f)*k_a[c0]);
  pk[bp+160+lane] = kr1*(1.0f + (a1 - 1.0f)*k_a[c1]);
  pk[bp+192+lane] = -kn0;            pk[bp+224+lane] = -kn1;
  pk[bp+256+lane] = kn0*a0;          pk[bp+288+lane] = kn1*a1;
  pk[bp+320+lane] = vr0 + (vf0 - vr0)*vs0;
  pk[bp+352+lane] = vr1 + (vf1 - vr1)*vs1;
}

// ---------------- sequential RWKV-7 scan ----------------
// State S[64] in VGPRs; the 1536B per-step record is double-buffered in LDS via
// async global->LDS copies issued one step ahead (latency hidden behind FMA chain).
__global__ __launch_bounds__(64)
void k_scan(float* __restrict__ pk, float* __restrict__ y)
{
  int bh = blockIdx.x; int b = bh / Hn, h = bh % Hn;
  int i = threadIdx.x;                       // value-row index, 0..63
  __shared__ __align__(16) float pb[2][384];
  float S[64];
#pragma unroll
  for (int j = 0; j < 64; ++j) S[j] = 0.0f;

  float* g0 = pk + ((size_t)(b*Tlen)*Hn + h)*384;
  async_ld16(g0 + i*4, &pb[0][i*4]);
  if (i < 32) async_ld16(g0 + 256 + i*4, &pb[0][256 + i*4]);
  __builtin_amdgcn_s_wait_asynccnt(0);
  __syncthreads();

  for (int t = 0; t < Tlen; ++t){
    int cur = t & 1, nxt = cur ^ 1;
    if (t + 1 < Tlen){
      float* gn = pk + ((size_t)(b*Tlen + t + 1)*Hn + h)*384;
      async_ld16(gn + i*4, &pb[nxt][i*4]);
      if (i < 32) async_ld16(gn + 256 + i*4, &pb[nxt][256 + i*4]);
    }
    const float* sd = &pb[cur][0];
    const float* sq = &pb[cur][64];
    const float* sk = &pb[cur][128];
    const float* sa = &pb[cur][192];
    const float* sb = &pb[cur][256];
    float vi = pb[cur][320 + i];
    float sacc = 0.0f;
#pragma unroll
    for (int j = 0; j < 64; ++j) sacc += S[j]*sa[j];
    float yi = 0.0f;
#pragma unroll
    for (int j = 0; j < 64; ++j){
      S[j] = S[j]*sd[j] + sacc*sb[j] + vi*sk[j];
      yi  += S[j]*sq[j];
    }
    y[((size_t)(b*Tlen + t))*Cdim + h*Nhd + i] = yi;
    __builtin_amdgcn_s_wait_asynccnt(0);   // next-step record landed in LDS
    __syncthreads();                        // visible to both waves; cur reads done
  }
}

// ---------------- GroupNorm + (r*k*r_k)·v residual + g gating -> f16 for final GEMM ----------------
__global__ __launch_bounds__(256)
void k_post(const float* __restrict__ y, const float* __restrict__ pk,
            const float* __restrict__ g, const float* __restrict__ r_k,
            const float* __restrict__ ln_w, const float* __restrict__ ln_b,
            _Float16* __restrict__ gy, int heads)
{
  int w = blockIdx.x*8 + (threadIdx.x >> 5);
  int lane = threadIdx.x & 31;
  if (w >= heads) return;
  int m = w / Hn, h = w % Hn;
  size_t bp = (size_t)w*384;
  size_t base = (size_t)m*Cdim + h*Nhd;
  int c0 = h*Nhd + lane, c1 = c0 + 32;
  size_t i0 = base + lane, i1 = i0 + 32;

  float y0 = y[i0], y1 = y[i1];
  float r0 = pk[bp+64+lane],  r1 = pk[bp+96+lane];
  float k0 = pk[bp+128+lane], k1 = pk[bp+160+lane];
  float v0 = pk[bp+320+lane], v1 = pk[bp+352+lane];
  float s1 = y0 + y1, s2 = y0*y0 + y1*y1;
  float dk = r0*k0*r_k[c0] + r1*k1*r_k[c1];
#pragma unroll
  for (int mm = 16; mm >= 1; mm >>= 1){
    s1 += __shfl_xor(s1, mm, 32);
    s2 += __shfl_xor(s2, mm, 32);
    dk += __shfl_xor(dk, mm, 32);
  }
  float mu  = s1*(1.0f/64.0f);
  float var = s2*(1.0f/64.0f) - mu*mu;
  float inv = rsqrtf(var + 6.4e-4f);   // EPS_GN = 1e-5 * 8 * 8
  float o0 = ((y0 - mu)*inv*ln_w[c0] + ln_b[c0] + dk*v0) * g[i0];
  float o1 = ((y1 - mu)*inv*ln_w[c1] + ln_b[c1] + dk*v1) * g[i1];
  gy[i0] = (_Float16)o0;  gy[i1] = (_Float16)o1;
}

// ---------------- host orchestration ----------------
extern "C" void kernel_launch(void* const* d_in, const int* in_sizes, int n_in,
                              void* d_out, int out_size, void* d_ws, size_t ws_size,
                              hipStream_t stream)
{
  (void)in_sizes; (void)n_in; (void)out_size; (void)ws_size;
  const float* x      = (const float*)d_in[0];
  const float* vfirst = (const float*)d_in[1];
  const float* mx_r   = (const float*)d_in[2];
  const float* mx_w   = (const float*)d_in[3];
  const float* mx_k   = (const float*)d_in[4];
  const float* mx_v   = (const float*)d_in[5];
  const float* mx_a   = (const float*)d_in[6];
  const float* mx_g   = (const float*)d_in[7];
  const float* w0     = (const float*)d_in[8];
  const float* w1     = (const float*)d_in[9];
  const float* w2     = (const float*)d_in[10];
  const float* a0     = (const float*)d_in[11];
  const float* a1     = (const float*)d_in[12];
  const float* a2     = (const float*)d_in[13];
  const float* v0     = (const float*)d_in[14];
  const float* v1     = (const float*)d_in[15];
  const float* v2     = (const float*)d_in[16];
  const float* g1     = (const float*)d_in[17];
  const float* g2     = (const float*)d_in[18];
  const float* k_k    = (const float*)d_in[19];
  const float* k_a    = (const float*)d_in[20];
  const float* r_k    = (const float*)d_in[21];
  const float* Wr     = (const float*)d_in[22];
  const float* Wk     = (const float*)d_in[23];
  const float* Wv     = (const float*)d_in[24];
  const float* Wo     = (const float*)d_in[25];
  const float* ln_w   = (const float*)d_in[26];
  const float* ln_b   = (const float*)d_in[27];
  float* out = (float*)d_out;

  const int M = Mrows, C = Cdim;
  const size_t MC = (size_t)M*C;

  char* ws = (char*)d_ws;
  size_t off = 0;
  auto alloc = [&](size_t bytes)->char*{ char* p = ws + off; off += (bytes + 255) & ~(size_t)255; return p; };

  _Float16* hAct = (_Float16*)alloc(MC*2);                 // reused: xr/xw/xk/xv/xa/xg, then gy
  _Float16* hHw  = (_Float16*)alloc((size_t)M*96*2);
  _Float16* hHa  = (_Float16*)alloc((size_t)M*96*2);
  _Float16* hHv  = (_Float16*)alloc((size_t)M*64*2);
  _Float16* hHg  = (_Float16*)alloc((size_t)M*256*2);
  _Float16* hWr  = (_Float16*)alloc((size_t)C*C*2);
  _Float16* hWk  = (_Float16*)alloc((size_t)C*C*2);
  _Float16* hWv  = (_Float16*)alloc((size_t)C*C*2);
  _Float16* hWo  = (_Float16*)alloc((size_t)C*C*2);
  _Float16* hW1t = (_Float16*)alloc((size_t)96*C*2);
  _Float16* hW2t = (_Float16*)alloc((size_t)C*96*2);
  _Float16* hA1t = (_Float16*)alloc((size_t)96*C*2);
  _Float16* hA2t = (_Float16*)alloc((size_t)C*96*2);
  _Float16* hV1t = (_Float16*)alloc((size_t)64*C*2);
  _Float16* hV2t = (_Float16*)alloc((size_t)C*64*2);
  _Float16* hG1t = (_Float16*)alloc((size_t)256*C*2);
  _Float16* hG2t = (_Float16*)alloc((size_t)C*256*2);
  float* fPK = (float*)alloc((size_t)M*PKROW*4);   // packed scan records (replaces 6 buffers)
  float* fG  = (float*)alloc(MC*4);
  float* fY  = (float*)alloc(MC*4);

  auto gemm = [&](const _Float16* Ap, const _Float16* Bp, int Mm, int Nn, int Kk,
                  float* oF, _Float16* oH, const float* bias, int mode, int pslot){
    int strips = Nn / 64;                   // full 64-wide strips (NT=4)
    dim3 gmain(strips, (Mm + 255)/256);
    if (strips > 0)
      k_gemm_wmma<4><<<gmain, 256, 0, stream>>>(Ap, Bp, Mm, Nn, Kk, 0, oF, oH, bias, mode, pslot);
    int rem = (Nn - strips*64) / 16;        // leftover 16-wide tiles (N=96 -> 2)
    if (rem > 0){
      dim3 gtail(1, (Mm + 255)/256);
      int nb = strips*64;
      switch (rem){
        case 1: k_gemm_wmma<1><<<gtail, 256, 0, stream>>>(Ap, Bp, Mm, Nn, Kk, nb, oF, oH, bias, mode, pslot); break;
        case 2: k_gemm_wmma<2><<<gtail, 256, 0, stream>>>(Ap, Bp, Mm, Nn, Kk, nb, oF, oH, bias, mode, pslot); break;
        default: k_gemm_wmma<3><<<gtail, 256, 0, stream>>>(Ap, Bp, Mm, Nn, Kk, nb, oF, oH, bias, mode, pslot); break;
      }
    }
  };
  auto mix = [&](const float* mv){
    k_mix<<<((int)MC + 255)/256, 256, 0, stream>>>(x, mv, hAct, Tlen, C, (int)MC);
  };

  // ---- weight conversion (f32 -> f16, low-rank transposed to [N][K]) ----
  int cc = C*C;
  k_convert_f16<<<(cc+255)/256, 256, 0, stream>>>(Wr, hWr, cc);
  k_convert_f16<<<(cc+255)/256, 256, 0, stream>>>(Wk, hWk, cc);
  k_convert_f16<<<(cc+255)/256, 256, 0, stream>>>(Wv, hWv, cc);
  k_convert_f16<<<(cc+255)/256, 256, 0, stream>>>(Wo, hWo, cc);
  k_transpose_f16<<<(C*96 +255)/256, 256, 0, stream>>>(w1, hW1t, C,  96);
  k_transpose_f16<<<(C*96 +255)/256, 256, 0, stream>>>(w2, hW2t, 96, C );
  k_transpose_f16<<<(C*96 +255)/256, 256, 0, stream>>>(a1, hA1t, C,  96);
  k_transpose_f16<<<(C*96 +255)/256, 256, 0, stream>>>(a2, hA2t, 96, C );
  k_transpose_f16<<<(C*64 +255)/256, 256, 0, stream>>>(v1, hV1t, C,  64);
  k_transpose_f16<<<(C*64 +255)/256, 256, 0, stream>>>(v2, hV2t, 64, C );
  k_transpose_f16<<<(C*256+255)/256, 256, 0, stream>>>(g1, hG1t, C,  256);
  k_transpose_f16<<<(C*256+255)/256, 256, 0, stream>>>(g2, hG2t, 256, C);

  // ---- projections (mix -> GEMM chains); scan operands land in packed layout ----
  mix(mx_r); gemm(hAct, hWr,  M, C,   C,   fPK,  nullptr, nullptr, MODE_F32,       1);
  mix(mx_w); gemm(hAct, hW1t, M, 96,  C,   nullptr, hHw,  nullptr, MODE_TANH_F16, -1);
             gemm(hHw,  hW2t, M, C,   96,  fPK,  nullptr, w0,      MODE_W_DECAY_F32, 0);
  mix(mx_k); gemm(hAct, hWk,  M, C,   C,   fPK,  nullptr, nullptr, MODE_F32,       2);
  mix(mx_v); gemm(hAct, hWv,  M, C,   C,   fPK,  nullptr, nullptr, MODE_F32,       5);
             gemm(hAct, hV1t, M, 64,  C,   nullptr, hHv,  nullptr, MODE_F16,      -1);
             gemm(hHv,  hV2t, M, C,   64,  fPK,  nullptr, v0,      MODE_BIAS_SIG_F32, 4);
  mix(mx_a); gemm(hAct, hA1t, M, 96,  C,   nullptr, hHa,  nullptr, MODE_F16,      -1);
             gemm(hHa,  hA2t, M, C,   96,  fPK,  nullptr, a0,      MODE_BIAS_SIG_F32, 3);
  mix(mx_g); gemm(hAct, hG1t, M, 256, C,   nullptr, hHg,  nullptr, MODE_SIG_F16,  -1);
             gemm(hHg,  hG2t, M, C,   256, fG,   nullptr, nullptr, MODE_F32,      -1);

  // ---- per-head prep, sequential scan (async LDS pipeline), post ----
  int heads = M*Hn;
  k_prep<<<(heads+7)/8, 256, 0, stream>>>(fPK, vfirst, k_k, k_a, heads);
  k_scan<<<Bsz*Hn, 64, 0, stream>>>(fPK, fY);
  k_post<<<(heads+7)/8, 256, 0, stream>>>(fY, fPK, fG, r_k, ln_w, ln_b, hAct, heads);

  // ---- output projection ----
  gemm(hAct, hWo, M, C, C, out, nullptr, nullptr, MODE_F32, -1);
}
